// GATEncoder_32255204393508
// MI455X (gfx1250) — matrix operand order repaired and checked by reading, not compiled, not run on hardware
//
#include <hip/hip_runtime.h>
#include <hip/hip_bf16.h>
#include <math.h>

// ---------------- problem constants (match reference) ----------------
#define NN   100000      // nodes
#define EE   1600000     // edges (before self-loops)
#define ET   (EE + NN)   // edges incl. self-loops
#define LRS  0.2f        // leaky relu slope

typedef __attribute__((ext_vector_type(16))) __bf16 v16bf;
typedef __attribute__((ext_vector_type(8)))  float  v8f;

// ---------------- helpers ----------------
__device__ __forceinline__ float lrelu(float x) { return x > 0.f ? x : LRS * x; }

// monotone float<->uint mapping so unsigned atomicMax == float max
__device__ __forceinline__ unsigned f2ou(float f) {
  unsigned u = __float_as_uint(f);
  return (u & 0x80000000u) ? ~u : (u | 0x80000000u);
}
__device__ __forceinline__ float ou2f(unsigned u) {
  return __uint_as_float((u & 0x80000000u) ? (u & 0x7FFFFFFFu) : ~u);
}

__device__ __forceinline__ void edge_sd(const int* __restrict__ ei, int e, int& s, int& d) {
  if (e < EE) { s = ei[e]; d = ei[EE + e]; }
  else        { s = e - EE; d = e - EE; }     // self loop
}

// ---------------- bf16 WMMA GEMM:  Y[nrows,NOUT] = X[nrows,K] @ W[K,NOUT] ----------------
// 256 threads = 8 waves; wave w handles rows [blk*128 + w*16, +16), all NOUT cols.
// A-fragment (16-bit A 16x32): lane(half=lane>>4, m=lane&15),
//   elements j -> K = kb + half*16 + j
// B-fragment (16-bit B KxN striping): elements j -> K = kb + half*16 + j, col = lane&15
template <int K, int NOUT>
__global__ __launch_bounds__(256) void gat_gemm_wmma(
    const float* __restrict__ X, const float* __restrict__ W,
    float* __restrict__ Y, int nrows) {
  constexpr int KP = K + 8;          // padded LDS stride (bf16 elems)
  constexpr int NT = NOUT / 16;      // N tiles
  constexpr int KS = K / 32;         // K steps
  __shared__ __bf16 Wl[NOUT * KP];   // W transposed: Wl[n*KP + k]

  const int t = threadIdx.x;
  for (int idx = t; idx < K * NOUT; idx += 256) {   // K*NOUT is multiple of 256
    int k = idx / NOUT, n = idx - k * NOUT;
    Wl[n * KP + k] = (__bf16)W[idx];
  }
  __syncthreads();

  const int wave = t >> 5, lane = t & 31;
  const int r16 = lane & 15, half = lane >> 4;
  const int rowBase = blockIdx.x * 128 + wave * 16;
  const int rowc = min(rowBase + r16, nrows - 1);   // clamped load row (no divergence)

  v16bf afrag[KS];
#pragma unroll
  for (int kk = 0; kk < KS; ++kk) {
    const float* p = X + (size_t)rowc * K + kk * 32 + half * 16;
#pragma unroll
    for (int j = 0; j < 8; ++j) {
      afrag[kk][j]     = (__bf16)p[j];
      afrag[kk][j + 8] = (__bf16)p[j + 8];
    }
  }

  v8f acc[NT] = {};
#pragma unroll
  for (int kk = 0; kk < KS; ++kk) {
#pragma unroll
    for (int nt = 0; nt < NT; ++nt) {
      const int col = nt * 16 + r16;
      const __bf16* q = &Wl[col * KP + kk * 32 + half * 16];
      v16bf bfrag;
#pragma unroll
      for (int j = 0; j < 16; ++j) bfrag[j] = q[j];
      acc[nt] = __builtin_amdgcn_wmma_f32_16x16x32_bf16(
          false, afrag[kk], false, bfrag, (short)0, acc[nt], false, false);
    }
  }

  // D layout: VGPR r -> row = rowBase + half*8 + r, col = N-tile base + (lane&15)
  const int orow = rowBase + half * 8;
#pragma unroll
  for (int nt = 0; nt < NT; ++nt) {
    const int col = nt * 16 + r16;
#pragma unroll
    for (int r = 0; r < 8; ++r) {
      const int rr = orow + r;
      if (rr < nrows) Y[(size_t)rr * NOUT + col] = acc[nt][r];
    }
  }
}

// ---------------- per-node attention logits ----------------
__global__ void gat_logits1(const float* __restrict__ h, const float* __restrict__ as,
                            const float* __restrict__ ad, float* __restrict__ alsrc,
                            float* __restrict__ aldst) {
  int n = blockIdx.x * blockDim.x + threadIdx.x;
  if (n >= NN) return;
  const float* hp = h + (size_t)n * 64;
#pragma unroll
  for (int hh = 0; hh < 8; ++hh) {
    float s = 0.f, d = 0.f;
#pragma unroll
    for (int c = 0; c < 8; ++c) {
      float v = hp[hh * 8 + c];
      s += v * as[hh * 8 + c];
      d += v * ad[hh * 8 + c];
    }
    alsrc[n * 8 + hh] = s;
    aldst[n * 8 + hh] = d;
  }
}

__global__ void gat_logits2(const float* __restrict__ h, const float* __restrict__ as,
                            const float* __restrict__ ad, float* __restrict__ alsrc,
                            float* __restrict__ aldst) {
  int n = blockIdx.x * blockDim.x + threadIdx.x;
  if (n >= NN) return;
  const float* hp = h + (size_t)n * 128;
  float s = 0.f, d = 0.f;
#pragma unroll 4
  for (int c = 0; c < 128; ++c) {
    float v = hp[c];
    s += v * as[c];
    d += v * ad[c];
  }
  alsrc[n] = s;
  aldst[n] = d;
}

// ---------------- layer-1 edge passes (H=8) ----------------
__global__ void gat_emax1(const int* __restrict__ ei, const float* __restrict__ alsrc,
                          const float* __restrict__ aldst, unsigned* __restrict__ mu) {
  int i = blockIdx.x * blockDim.x + threadIdx.x;
  if (i >= ET * 8) return;
  int e = i >> 3, hh = i & 7, s, d;
  edge_sd(ei, e, s, d);
  float x = lrelu(alsrc[s * 8 + hh] + aldst[d * 8 + hh]);
  atomicMax(&mu[d * 8 + hh], f2ou(x));
}

__global__ void gat_esum1(const int* __restrict__ ei, const float* __restrict__ alsrc,
                          const float* __restrict__ aldst, const unsigned* __restrict__ mu,
                          float* __restrict__ denom) {
  int i = blockIdx.x * blockDim.x + threadIdx.x;
  if (i >= ET * 8) return;
  int e = i >> 3, hh = i & 7, s, d;
  edge_sd(ei, e, s, d);
  float x = lrelu(alsrc[s * 8 + hh] + aldst[d * 8 + hh]);
  atomicAdd(&denom[d * 8 + hh], expf(x - ou2f(mu[d * 8 + hh])));
}

__global__ void gat_escatter1(const int* __restrict__ ei, const float* __restrict__ alsrc,
                              const float* __restrict__ aldst, const unsigned* __restrict__ mu,
                              const float* __restrict__ denom, const float* __restrict__ h,
                              float* __restrict__ agg) {
  int i = blockIdx.x * blockDim.x + threadIdx.x;
  if (i >= ET * 8) return;
  int e = i >> 3, hh = i & 7, s, d;
  edge_sd(ei, e, s, d);
  float x = lrelu(alsrc[s * 8 + hh] + aldst[d * 8 + hh]);
  float alpha = expf(x - ou2f(mu[d * 8 + hh])) / (denom[d * 8 + hh] + 1e-16f);
  const float4* hp = (const float4*)(h + (size_t)s * 64 + hh * 8);
  float* op = agg + (size_t)d * 64 + hh * 8;
  float4 v0 = hp[0], v1 = hp[1];
  atomicAdd(&op[0], v0.x * alpha); atomicAdd(&op[1], v0.y * alpha);
  atomicAdd(&op[2], v0.z * alpha); atomicAdd(&op[3], v0.w * alpha);
  atomicAdd(&op[4], v1.x * alpha); atomicAdd(&op[5], v1.y * alpha);
  atomicAdd(&op[6], v1.z * alpha); atomicAdd(&op[7], v1.w * alpha);
}

// agg += b1, then ELU, in place (feeds layer-2 GEMM)
__global__ void gat_bias_elu(float* __restrict__ agg, const float* __restrict__ b1) {
  int i = blockIdx.x * blockDim.x + threadIdx.x;
  if (i >= NN * 64) return;
  float v = agg[i] + b1[i & 63];
  agg[i] = v > 0.f ? v : expm1f(v);
}

// ---------------- layer-2 edge passes (H=1, C=128) ----------------
__global__ void gat_emax2(const int* __restrict__ ei, const float* __restrict__ alsrc,
                          const float* __restrict__ aldst, unsigned* __restrict__ mu) {
  int e = blockIdx.x * blockDim.x + threadIdx.x;
  if (e >= ET) return;
  int s, d;
  edge_sd(ei, e, s, d);
  atomicMax(&mu[d], f2ou(lrelu(alsrc[s] + aldst[d])));
}

__global__ void gat_esum2(const int* __restrict__ ei, const float* __restrict__ alsrc,
                          const float* __restrict__ aldst, const unsigned* __restrict__ mu,
                          float* __restrict__ denom) {
  int e = blockIdx.x * blockDim.x + threadIdx.x;
  if (e >= ET) return;
  int s, d;
  edge_sd(ei, e, s, d);
  float x = lrelu(alsrc[s] + aldst[d]);
  atomicAdd(&denom[d], expf(x - ou2f(mu[d])));
}

// 8 threads per edge, 16 channels each (coalesced 64B chunks of h2[src])
__global__ void gat_escatter2(const int* __restrict__ ei, const float* __restrict__ alsrc,
                              const float* __restrict__ aldst, const unsigned* __restrict__ mu,
                              const float* __restrict__ denom, const float* __restrict__ h,
                              float* __restrict__ out) {
  int i = blockIdx.x * blockDim.x + threadIdx.x;
  if (i >= ET * 8) return;
  int e = i >> 3, sub = i & 7, s, d;
  edge_sd(ei, e, s, d);
  float x = lrelu(alsrc[s] + aldst[d]);
  float alpha = expf(x - ou2f(mu[d])) / (denom[d] + 1e-16f);
  const float4* hp = (const float4*)(h + (size_t)s * 128 + sub * 16);
  float* op = out + (size_t)d * 128 + sub * 16;
#pragma unroll
  for (int q = 0; q < 4; ++q) {
    float4 v = hp[q];
    atomicAdd(&op[q * 4 + 0], v.x * alpha);
    atomicAdd(&op[q * 4 + 1], v.y * alpha);
    atomicAdd(&op[q * 4 + 2], v.z * alpha);
    atomicAdd(&op[q * 4 + 3], v.w * alpha);
  }
}

// ---------------- init kernels ----------------
__global__ void gat_zero(float* __restrict__ p, int n) {
  int i = blockIdx.x * blockDim.x + threadIdx.x;
  if (i < n) p[i] = 0.f;
}
__global__ void gat_init_out(float* __restrict__ out, const float* __restrict__ b2) {
  int i = blockIdx.x * blockDim.x + threadIdx.x;
  if (i < NN * 128) out[i] = b2[i & 127];
}

// ---------------- launch ----------------
extern "C" void kernel_launch(void* const* d_in, const int* in_sizes, int n_in,
                              void* d_out, int out_size, void* d_ws, size_t ws_size,
                              hipStream_t stream) {
  const float* x   = (const float*)d_in[0];
  const int*   ei  = (const int*)  d_in[1];
  const float* W1  = (const float*)d_in[2];
  const float* as1 = (const float*)d_in[3];
  const float* ad1 = (const float*)d_in[4];
  const float* b1  = (const float*)d_in[5];
  const float* W2  = (const float*)d_in[6];
  const float* as2 = (const float*)d_in[7];
  const float* ad2 = (const float*)d_in[8];
  const float* b2  = (const float*)d_in[9];
  float* out = (float*)d_out;

  // workspace layout (floats); zero-init region is contiguous at the end
  float* ws = (float*)d_ws;
  size_t o = 0;
  float*    h1     = ws + o; o += (size_t)NN * 64;   // x @ W1
  float*    alsrc1 = ws + o; o += (size_t)NN * 8;
  float*    aldst1 = ws + o; o += (size_t)NN * 8;
  float*    h2     = ws + o; o += (size_t)NN * 128;  // elu(agg1+b1) @ W2
  float*    alsrc2 = ws + o; o += (size_t)NN;
  float*    aldst2 = ws + o; o += (size_t)NN;
  float*    zbase  = ws + o;                         // ---- zeroed every call ----
  unsigned* m1u    = (unsigned*)(ws + o); o += (size_t)NN * 8;
  float*    den1   = ws + o; o += (size_t)NN * 8;
  float*    agg1   = ws + o; o += (size_t)NN * 64;
  unsigned* m2u    = (unsigned*)(ws + o); o += (size_t)NN;
  float*    den2   = ws + o; o += (size_t)NN;
  const int nzero  = NN * 82;                        // m1u..den2 contiguous
  (void)ws_size; (void)in_sizes; (void)n_in; (void)out_size;

  const int BS = 256;
  const int gN    = (NN + BS - 1) / BS;
  const int gRows = (NN + 127) / 128;
  const int gE    = (ET + BS - 1) / BS;
  const int gE8   = (ET * 8 + BS - 1) / BS;

  // ---- layer 1 ----
  gat_gemm_wmma<128, 64><<<gRows, BS, 0, stream>>>(x, W1, h1, NN);
  gat_logits1<<<gN, BS, 0, stream>>>(h1, as1, ad1, alsrc1, aldst1);
  gat_zero<<<(nzero + BS - 1) / BS, BS, 0, stream>>>(zbase, nzero);
  gat_emax1<<<gE8, BS, 0, stream>>>(ei, alsrc1, aldst1, m1u);
  gat_esum1<<<gE8, BS, 0, stream>>>(ei, alsrc1, aldst1, m1u, den1);
  gat_escatter1<<<gE8, BS, 0, stream>>>(ei, alsrc1, aldst1, m1u, den1, h1, agg1);
  gat_bias_elu<<<(NN * 64 + BS - 1) / BS, BS, 0, stream>>>(agg1, b1);

  // ---- layer 2 ----
  gat_gemm_wmma<64, 128><<<gRows, BS, 0, stream>>>(agg1, W2, h2, NN);
  gat_logits2<<<gN, BS, 0, stream>>>(h2, as2, ad2, alsrc2, aldst2);
  gat_init_out<<<(NN * 128 + BS - 1) / BS, BS, 0, stream>>>(out, b2);
  gat_emax2<<<gE, BS, 0, stream>>>(ei, alsrc2, aldst2, m2u);
  gat_esum2<<<gE, BS, 0, stream>>>(ei, alsrc2, aldst2, m2u, den2);
  gat_escatter2<<<gE8, BS, 0, stream>>>(ei, alsrc2, aldst2, m2u, den2, h2, out);
}
